// MambaLayer_824633721547
// MI455X (gfx1250) — compile-verified, matrix operand order checked by reading, not run on hardware
//
#include <hip/hip_runtime.h>
#include <hip/hip_bf16.h>
#include <math.h>

// ---------------- types ----------------
typedef __attribute__((ext_vector_type(16))) _Float16 v16h;
typedef __attribute__((ext_vector_type(8)))  _Float16 v8h;
typedef __attribute__((ext_vector_type(8)))  float    v8f;
typedef __attribute__((ext_vector_type(4)))  int      i32x4;
typedef __attribute__((address_space(1)))    i32x4    gi32x4;   // global int4
typedef __attribute__((address_space(3)))    i32x4    li32x4;   // LDS int4

union Frag16 { v16h v; v8h h[2]; };

#if defined(__AMDGCN__) && __has_builtin(__builtin_amdgcn_global_load_async_to_lds_b128) && __has_builtin(__builtin_amdgcn_s_wait_asynccnt)
#define USE_ASYNC_LDS 1
#else
#define USE_ASYNC_LDS 0
#endif

// ---------------- problem constants ----------------
#define BATCH   4
#define SEQ     4096
#define DMODEL  1024
#define DINNER  2048
#define DSTATE  16
#define DCONV   4
#define MROWS   (BATCH * SEQ)        // 16384

// ---------------- global fragment loader (row-major, stride K halves) ----------------
// 16-bit WMMA A/B fragment per CDNA5 ISA 16x32 layout:
// element e -> K = kbase + (e&7) + (e&8 ? 16 : 0) + (lane&16 ? 8 : 0)
// => two 16-byte vectors at +0 and +16 halves.
__device__ __forceinline__ v16h load_frag16_g(const _Float16* __restrict__ base,
                                              int strideK, int row, int kbase,
                                              int lane) {
  const _Float16* p = base + (size_t)row * strideK + kbase + ((lane & 16) ? 8 : 0);
  Frag16 f;
  f.h[0] = *(const v8h*)(p);
  f.h[1] = *(const v8h*)(p + 16);
  return f.v;
}

// LDS fragment loader: B tile stored as [col][64 halves] (col stride 64 halves).
__device__ __forceinline__ v16h load_frag16_lds(const _Float16* ldsBase,
                                                int col, int koff, int lane) {
  const _Float16* p = ldsBase + col * 64 + koff + ((lane & 16) ? 8 : 0);
  Frag16 f;
  f.h[0] = *(const v8h*)(p);
  f.h[1] = *(const v8h*)(p + 16);
  return f.v;
}

// ---------------- WMMA GEMM with LDS-staged B, double-buffered ----------------
// C[M,N] = A[M,K](f16 row-major) * Bt[N,K]^T (f16, stored N-by-K) + bias
// Block: 256 threads = 8 waves. Block tile: 256(M) x NT*16(N).
// Wave w owns rows [mBase, mBase+32) as 2 m-tiles; NT n-tiles; 2*NT accumulators.
// K consumed in 64-half chunks (2 wmma substeps); B chunk staged in LDS via
// GLOBAL_LOAD_ASYNC_TO_LDS_B128 (ASYNCcnt) when available, else via registers.
// mode 0: out0 = xi (f32, cols < DINNER), out1 = res (f16, cols >= DINNER)
// mode 1/2: out0 = f32, row stride N.
template <int NT>
__global__ void __launch_bounds__(256)
wmma_gemm_lds(const _Float16* __restrict__ A,
              const _Float16* __restrict__ Bt,
              const float* __restrict__ bias,
              void* __restrict__ out0, void* __restrict__ out1,
              int M, int N, int K, int mode) {
  constexpr int NSEG = NT * 16 * 8;          // b128 segments per B chunk
  constexpr int SPT  = (NSEG + 255) / 256;   // segments per thread (1 or 2)
  __shared__ _Float16 Bs[2][NT * 16 * 64];   // [buf][col*64 + khalf]

  const int tid   = threadIdx.x;
  const int lane  = tid & 31;
  const int wave  = tid >> 5;
  const int mBase = blockIdx.y * 256 + wave * 32;
  const int nBase = blockIdx.x * (NT * 16);
  const int row0  = mBase + (lane & 15);
  const int row1  = row0 + 16;

  v8f acc[2][NT];
#pragma unroll
  for (int mt = 0; mt < 2; ++mt)
#pragma unroll
    for (int t = 0; t < NT; ++t)
      acc[mt][t] = (v8f){};

  // ---- prologue: stage chunk k=0 into Bs[0] ----
#if USE_ASYNC_LDS
#pragma unroll
  for (int i = tid; i < NSEG; i += 256) {
    int r = i >> 3, s = i & 7;
    const _Float16* src = Bt + (size_t)(nBase + r) * K + s * 8;
    __builtin_amdgcn_global_load_async_to_lds_b128(
        (gi32x4*)(void*)src,
        (li32x4*)(void*)&Bs[0][r * 64 + s * 8],
        0, 0);
  }
  __builtin_amdgcn_s_wait_asynccnt(0);
#else
#pragma unroll
  for (int i = tid; i < NSEG; i += 256) {
    int r = i >> 3, s = i & 7;
    v8h v = *(const v8h*)(Bt + (size_t)(nBase + r) * K + s * 8);
    *(v8h*)(&Bs[0][r * 64 + s * 8]) = v;
  }
#endif
  __syncthreads();

  for (int k = 0; k < K; k += 64) {
    const int cur = (k >> 6) & 1;
    const bool have_next = (k + 64) < K;

    // prefetch A two chunks ahead (global_prefetch_b8)
    if (k + 128 < K)
      __builtin_prefetch(A + (size_t)row0 * K + k + 128, 0, 3);

    // stage the NEXT B chunk while this chunk computes
#if USE_ASYNC_LDS
    if (have_next) {
#pragma unroll
      for (int i = tid; i < NSEG; i += 256) {
        int r = i >> 3, s = i & 7;
        const _Float16* src = Bt + (size_t)(nBase + r) * K + (k + 64) + s * 8;
        __builtin_amdgcn_global_load_async_to_lds_b128(
            (gi32x4*)(void*)src,
            (li32x4*)(void*)&Bs[cur ^ 1][r * 64 + s * 8],
            0, 0);
      }
    }
#else
    v8h nreg[SPT];
    if (have_next) {
#pragma unroll
      for (int i = tid, c = 0; i < NSEG; i += 256, ++c) {
        int r = i >> 3, s = i & 7;
        nreg[c] = *(const v8h*)(Bt + (size_t)(nBase + r) * K + (k + 64) + s * 8);
      }
    }
#endif

    // A fragments for both 32-K substeps (global, per-wave rows)
    v16h a00 = load_frag16_g(A, K, row0, k,      lane);
    v16h a01 = load_frag16_g(A, K, row1, k,      lane);
    v16h a10 = load_frag16_g(A, K, row0, k + 32, lane);
    v16h a11 = load_frag16_g(A, K, row1, k + 32, lane);

    // compute: 2 substeps x NT n-tiles x 2 m-tiles
#pragma unroll
    for (int d = 0; d < 2; ++d) {
      const v16h am0 = d ? a10 : a00;
      const v16h am1 = d ? a11 : a01;
#pragma unroll
      for (int t = 0; t < NT; ++t) {
        v16h b = load_frag16_lds(&Bs[cur][0], t * 16 + (lane & 15), d * 32, lane);
        acc[0][t] = __builtin_amdgcn_wmma_f32_16x16x32_f16(
            false, am0, false, b, (short)0, acc[0][t], false, false);
        acc[1][t] = __builtin_amdgcn_wmma_f32_16x16x32_f16(
            false, am1, false, b, (short)0, acc[1][t], false, false);
      }
    }

#if USE_ASYNC_LDS
    __builtin_amdgcn_s_wait_asynccnt(0);
#else
    if (have_next) {
#pragma unroll
      for (int i = tid, c = 0; i < NSEG; i += 256, ++c) {
        int r = i >> 3, s = i & 7;
        *(v8h*)(&Bs[cur ^ 1][r * 64 + s * 8]) = nreg[c];
      }
    }
#endif
    __syncthreads();
  }

  // ---- epilogue ----
  const int moff = (lane & 16) ? 8 : 0;
#pragma unroll
  for (int t = 0; t < NT; ++t) {
    const int n  = nBase + t * 16 + (lane & 15);
    const float bv = bias ? bias[n] : 0.0f;
#pragma unroll
    for (int mt = 0; mt < 2; ++mt) {
#pragma unroll
      for (int r = 0; r < 8; ++r) {
        const int m = mBase + mt * 16 + r + moff;
        const float v = acc[mt][t][r] + bv;
        if (mode == 0) {
          if (n < DINNER)
            ((float*)out0)[(size_t)m * DINNER + n] = v;                        // xi f32
          else
            ((_Float16*)out1)[(size_t)m * DINNER + (n - DINNER)] = (_Float16)v; // res f16
        } else {
          ((float*)out0)[(size_t)m * N + n] = v;
        }
      }
    }
  }
}

// ---------------- elementwise: fp32 -> fp16 ----------------
__global__ void cvt_f32_to_f16(const float* __restrict__ in,
                               _Float16* __restrict__ out, int n) {
  int i = blockIdx.x * blockDim.x + threadIdx.x;
  if (i < n) out[i] = (_Float16)in[i];
}

// ---------------- transpose + convert: (R x C) f32 -> (C x R) f16 ----------------
__global__ void transpose_cvt(const float* __restrict__ in,
                              _Float16* __restrict__ out, int R, int C) {
  int i = blockIdx.x * blockDim.x + threadIdx.x;
  if (i < R * C) {
    int r = i / C, c = i - r * C;
    out[(size_t)c * R + r] = (_Float16)in[i];
  }
}

// ---------------- depthwise causal conv (width 4) + bias + SiLU -> f16 ----------------
__global__ void conv_silu_kernel(const float* __restrict__ xi,
                                 const float* __restrict__ cw,
                                 const float* __restrict__ cb,
                                 _Float16* __restrict__ xa) {
  int idx = blockIdx.x * blockDim.x + threadIdx.x;
  if (idx >= MROWS * DINNER) return;
  int m = idx >> 11;            // idx / DINNER
  int c = idx & (DINNER - 1);
  int b = m >> 12;              // m / SEQ
  int s = m & (SEQ - 1);
  float acc = cb[c];
#pragma unroll
  for (int t = 0; t < DCONV; ++t) {
    int ss = s - (DCONV - 1) + t;
    if (ss >= 0)
      acc += cw[c * DCONV + t] * xi[((size_t)(b * SEQ + ss) << 11) + c];
  }
  float v = acc / (1.0f + expf(-acc));   // SiLU
  xa[idx] = (_Float16)v;
}

// ---------------- sequential recurrence: one wave per batch ----------------
// h_t = tanh(Aexp[t%16] @ h_{t-1}) + B_t * h_{t-1} + C_t
__global__ void __launch_bounds__(32)
scan_kernel(const float* __restrict__ BmCm,   // [MROWS, 32] : [Bm(16) | Cm(16)]
            const float* __restrict__ dt_param,
            const float* __restrict__ Amat,   // [16,16]
            float* __restrict__ hs) {         // [MROWS, 16]
  __shared__ float Aexp[16 * 16 * 16];
  const int b    = blockIdx.x;
  const int lane = threadIdx.x;

  for (int i = lane; i < 16 * 16 * 16; i += 32) {
    int p = i >> 8;
    int j = (i >> 4) & 15;
    int k = i & 15;
    Aexp[i] = expf(expf(dt_param[p]) * Amat[j * 16 + k]);
  }
  __syncthreads();

  const int j = lane & 15;
  float h = 0.0f;
  const float* bc = BmCm + (size_t)b * SEQ * 32;
  float* hb = hs + (size_t)b * SEQ * 16;

  for (int t = 0; t < SEQ; ++t) {
    const int p = t & 15;
    const float Bt = bc[t * 32 + j];
    const float Ct = bc[t * 32 + 16 + j];
    const float* Ap = &Aexp[p * 256 + j * 16];
    float acc = 0.0f;
#pragma unroll
    for (int k = 0; k < 16; ++k) {
      float hk = __shfl(h, k, 32);   // lanes 0..15 hold the state
      acc += Ap[k] * hk;
    }
    float hn = tanhf(acc) + Bt * h + Ct;
    h = hn;
    if (lane < 16) hb[t * 16 + j] = hn;
  }
}

// ---------------- y = repeat(hs, 128) + res  (f16) ----------------
__global__ void build_y_kernel(const float* __restrict__ hs,
                               const _Float16* __restrict__ res,
                               _Float16* __restrict__ y) {
  int idx = blockIdx.x * blockDim.x + threadIdx.x;
  if (idx >= MROWS * DINNER) return;
  int m = idx >> 11;
  int c = idx & (DINNER - 1);
  float v = hs[(size_t)m * DSTATE + (c >> 7)] + (float)res[idx];
  y[idx] = (_Float16)v;
}

// ---------------- host launcher ----------------
extern "C" void kernel_launch(void* const* d_in, const int* in_sizes, int n_in,
                              void* d_out, int out_size, void* d_ws, size_t ws_size,
                              hipStream_t stream) {
  const float* x        = (const float*)d_in[0];
  const float* W_in     = (const float*)d_in[1];
  const float* b_in     = (const float*)d_in[2];
  const float* conv_w   = (const float*)d_in[3];
  const float* conv_b   = (const float*)d_in[4];
  const float* W_xp     = (const float*)d_in[5];
  const float* b_xp     = (const float*)d_in[6];
  const float* dt_param = (const float*)d_in[7];
  const float* Amat     = (const float*)d_in[8];
  // d_in[9] = Dp (unused by reference)
  const float* W_out    = (const float*)d_in[10];
  const float* b_out    = (const float*)d_in[11];
  float* out = (float*)d_out;

  // ---- workspace carve-out ----
  char* w = (char*)d_ws;
  size_t off = 0;
  auto take = [&](size_t bytes) {
    void* p = w + off;
    off += (bytes + 255) & ~(size_t)255;
    return p;
  };
  _Float16* x_h    = (_Float16*)take((size_t)MROWS * DMODEL * 2);        // x in f16
  _Float16* WinT   = (_Float16*)take((size_t)(2 * DINNER) * DMODEL * 2); // W_in^T (4096x1024)
  float*    xi     = (float*)   take((size_t)MROWS * DINNER * 4);        // pre-conv half (f32)
  _Float16* res_h  = (_Float16*)take((size_t)MROWS * DINNER * 2);        // residual half (f16)
  _Float16* xa_h   = (_Float16*)take((size_t)MROWS * DINNER * 2);        // post conv+silu (f16)
  _Float16* WxpT   = (_Float16*)take((size_t)(2 * DSTATE) * DINNER * 2); // W_xp^T (32x2048)
  float*    BmCm   = (float*)   take((size_t)MROWS * 2 * DSTATE * 4);    // [Bm|Cm]
  float*    hs     = (float*)   take((size_t)MROWS * DSTATE * 4);
  _Float16* y_h    = (_Float16*)take((size_t)MROWS * DINNER * 2);        // y (f16)
  _Float16* WoutT  = (_Float16*)take((size_t)DMODEL * DINNER * 2);       // W_out^T (1024x2048)

  const int T = 256;

  // 1) convert / transpose weights & activations to f16
  {
    int n = MROWS * DMODEL;
    cvt_f32_to_f16<<<(n + T - 1) / T, T, 0, stream>>>(x, x_h, n);
  }
  {
    int n = DMODEL * 2 * DINNER;
    transpose_cvt<<<(n + T - 1) / T, T, 0, stream>>>(W_in, WinT, DMODEL, 2 * DINNER);
  }
  {
    int n = DINNER * 2 * DSTATE;
    transpose_cvt<<<(n + T - 1) / T, T, 0, stream>>>(W_xp, WxpT, DINNER, 2 * DSTATE);
  }
  {
    int n = DINNER * DMODEL;
    transpose_cvt<<<(n + T - 1) / T, T, 0, stream>>>(W_out, WoutT, DINNER, DMODEL);
  }

  // 2) GEMM1: xr = x @ W_in + b_in  -> split into xi (f32) / res (f16)
  {
    dim3 grid((2 * DINNER) / 64, MROWS / 256);
    wmma_gemm_lds<4><<<grid, 256, 0, stream>>>(x_h, WinT, b_in, xi, res_h,
                                               MROWS, 2 * DINNER, DMODEL, 0);
  }

  // 3) depthwise causal conv + SiLU -> xa (f16)
  {
    int n = MROWS * DINNER;
    conv_silu_kernel<<<(n + T - 1) / T, T, 0, stream>>>(xi, conv_w, conv_b, xa_h);
  }

  // 4) GEMM2: xdbl = xa @ W_xp + b_xp -> [Bm|Cm] (f32), N = 32 (NT = 2)
  {
    dim3 grid(1, MROWS / 256);
    wmma_gemm_lds<2><<<grid, 256, 0, stream>>>(xa_h, WxpT, b_xp, BmCm, nullptr,
                                               MROWS, 2 * DSTATE, DINNER, 1);
  }

  // 5) serial recurrence: one wave per batch
  scan_kernel<<<BATCH, 32, 0, stream>>>(BmCm, dt_param, Amat, hs);

  // 6) y = repeat(hs, 128) + res  (f16)
  {
    int n = MROWS * DINNER;
    build_y_kernel<<<(n + T - 1) / T, T, 0, stream>>>(hs, res_h, y_h);
  }

  // 7) GEMM3: out = y @ W_out + b_out (f32)
  {
    dim3 grid(DMODEL / 64, MROWS / 256);
    wmma_gemm_lds<4><<<grid, 256, 0, stream>>>(y_h, WoutT, b_out, out, nullptr,
                                               MROWS, DMODEL, DINNER, 2);
  }
}
